// LiquidObserver_20048907337862
// MI455X (gfx1250) — matrix operand order repaired and checked by reading, not compiled, not run on hardware
//
#include <hip/hip_runtime.h>
#include <math.h>

// ---------------------------------------------------------------------------
// CDNA5 (gfx1250) wave32 WMMA implementation of the LiquidObserver model.
//   x -> bf16                                  (bandwidth pass)
//   proj = x @ w_in + b_in                     (bf16 WMMA GEMM, async-LDS A tiles)
//   scan over T=1024 of 3 CfC cells            (persistent, TDM-prefetched xt tiles)
//   out  = gelu(cfc @ hw1 + hb1) @ hw2 + hb2   (fused bf16 WMMA head)
// ---------------------------------------------------------------------------

typedef __attribute__((ext_vector_type(16))) __bf16 v16bf;
typedef __attribute__((ext_vector_type(8)))  float  v8f;
typedef unsigned int v4ui __attribute__((ext_vector_type(4)));
typedef int v8si __attribute__((ext_vector_type(8)));
typedef int v4si __attribute__((ext_vector_type(4)));
typedef int v4i_vs __attribute__((vector_size(16)));   // matches async builtin param

#define AS1 __attribute__((address_space(1)))
#define AS3 __attribute__((address_space(3)))

// Feature probes (device pass only; host pass always parses the fallback).
#if defined(__HIP_DEVICE_COMPILE__) && __has_builtin(__builtin_amdgcn_global_load_async_to_lds_b128)
#define HAVE_ASYNC 1
#else
#define HAVE_ASYNC 0
#endif
#if defined(__HIP_DEVICE_COMPILE__) && __has_builtin(__builtin_amdgcn_tensor_load_to_lds)
#define HAVE_TDM 1
#else
#define HAVE_TDM 0
#endif

#define B_DIM 64
#define T_STEPS 1024
#define IN_DIM 1024
#define H_DIM 512
#define OUT_DIM 256
#define M_ROWS (B_DIM * T_STEPS)   // 65536

// Segment-padded layer geometry (K padded to mult of 32, N to mult of 16)
#define CATP0 672
#define CATP1 288
#define CATP2 384
#define HP0 160
#define HP1 112
#define HP2 256

__device__ __forceinline__ v8f zero8() {
  v8f z;
#pragma unroll
  for (int i = 0; i < 8; ++i) z[i] = 0.f;
  return z;
}

// A fragment (16x32 bf16), ISA layout: lane<16 -> K 0..7 (v0-3) & 16..23 (v4-7),
// lane>=16 -> K 8..15 & 24..31. base points at row 0 of the 16-row tile.
__device__ __forceinline__ v16bf load_afrag(const __bf16* base, int ldk, int k0) {
  const int lane = threadIdx.x & 31;
  const __bf16* p = base + (size_t)(lane & 15) * ldk + k0 + ((lane >> 4) << 3);
  v16bf a;
#pragma unroll
  for (int i = 0; i < 8; ++i) { a[i] = p[i]; a[i + 8] = p[i + 16]; }
  return a;
}

// B fragment (32x16 bf16) from column-major (transposed) weights.
__device__ __forceinline__ v16bf load_bfrag(const __bf16* colp, int k0) {
  const int lane = threadIdx.x & 31;
  const __bf16* p = colp + k0 + ((lane >> 4) << 3);
  v16bf b;
#pragma unroll
  for (int i = 0; i < 8; ++i) { b[i] = p[i]; b[i + 8] = p[i + 16]; }
  return b;
}

__device__ __forceinline__ v8f wmma_bf16(v16bf a, v16bf b, v8f c) {
  return __builtin_amdgcn_wmma_f32_16x16x32_bf16(false, a, false, b, (short)0, c,
                                                 false, false);
}

__device__ __forceinline__ float sigmoidf_(float x) { return 1.f / (1.f + __expf(-x)); }

// 16-element (32B) bf16 copy global->LDS: async-to-LDS when available.
__device__ __forceinline__ void copy16_g2l(const __bf16* g, __bf16* l) {
#if HAVE_ASYNC
  __builtin_amdgcn_global_load_async_to_lds_b128((AS1 v4i_vs*)g, (AS3 v4i_vs*)l, 0, 0);
  __builtin_amdgcn_global_load_async_to_lds_b128((AS1 v4i_vs*)(g + 8), (AS3 v4i_vs*)(l + 8), 0, 0);
#else
  ((uint4*)l)[0] = ((const uint4*)g)[0];
  ((uint4*)l)[1] = ((const uint4*)g)[1];
#endif
}

__device__ __forceinline__ void wait_async2() {
#if HAVE_ASYNC
#if __has_builtin(__builtin_amdgcn_s_wait_asynccnt)
  __builtin_amdgcn_s_wait_asynccnt(2);
#else
  asm volatile("s_wait_asynccnt 0x2" ::: "memory");
#endif
#endif
}
__device__ __forceinline__ void wait_async0() {
#if HAVE_ASYNC
#if __has_builtin(__builtin_amdgcn_s_wait_asynccnt)
  __builtin_amdgcn_s_wait_asynccnt(0);
#else
  asm volatile("s_wait_asynccnt 0x0" ::: "memory");
#endif
#endif
}

__device__ __forceinline__ unsigned lds_addr_of(const void* p) {
  return (unsigned)(unsigned long long)(AS3 const void*)p;
}

#if HAVE_TDM
// Issue one TDM 2D tile load: tile_h rows x tile_w bf16 elements, row stride
// row_stride_elems, into LDS byte address lds_byte. D# bitfields per ISA 8.3/8.4.
__device__ __forceinline__ void tdm_load_tile(const __bf16* gsrc, unsigned lds_byte,
                                              unsigned tile_w, unsigned tile_h,
                                              unsigned long long row_stride_elems) {
  const unsigned long long ga = (unsigned long long)gsrc;
  v4ui g0;
  g0[0] = 1u;                                              // count=1, user desc
  g0[1] = lds_byte;                                        // lds_addr
  g0[2] = (unsigned)(ga & 0xFFFFFFFFu);                    // global_addr lo
  g0[3] = (unsigned)((ga >> 32) & 0x1FFFFFFu) | (2u << 30); // global_addr hi | type=2
  v8si g1;
  g1[0] = (int)(1u << 16);                                 // data_size = 2B
  g1[1] = (int)((tile_w & 0xFFFFu) << 16);                 // tensor_dim0 lo16
  g1[2] = (int)(((tile_w >> 16) & 0xFFFFu) | ((tile_h & 0xFFFFu) << 16)); // dim0 hi | dim1 lo
  g1[3] = (int)(((tile_h >> 16) & 0xFFFFu) | ((tile_w & 0xFFFFu) << 16)); // dim1 hi | tile_dim0
  g1[4] = (int)(tile_h & 0xFFFFu);                         // tile_dim1 (tile_dim2=0)
  g1[5] = (int)(row_stride_elems & 0xFFFFFFFFull);         // tensor_dim0_stride lo32
  g1[6] = (int)((row_stride_elems >> 32) & 0xFFFFull);     // stride hi16 (dim1_stride=0)
  g1[7] = 0;
  v4si gz = {0, 0, 0, 0};
#if __clang_major__ >= 23
  v8si gz8 = {0, 0, 0, 0, 0, 0, 0, 0};
  __builtin_amdgcn_tensor_load_to_lds(g0, g1, gz, gz, gz8, 0);
#else
  __builtin_amdgcn_tensor_load_to_lds(g0, g1, gz, gz, 0);
#endif
}
#endif

// ---------------------------------------------------------------------------
// f32 -> bf16 bulk convert (8 elems/thread)
// ---------------------------------------------------------------------------
__global__ __launch_bounds__(256) void k_cvt_bf16(const float* __restrict__ src,
                                                  __bf16* __restrict__ dst, size_t n) {
  size_t i = ((size_t)blockIdx.x * blockDim.x + threadIdx.x) * 8;
  if (i >= n) return;
  const float4 a = *(const float4*)(src + i);
  const float4 b = *(const float4*)(src + i + 4);
  __bf16 tmp[8] = {(__bf16)a.x, (__bf16)a.y, (__bf16)a.z, (__bf16)a.w,
                   (__bf16)b.x, (__bf16)b.y, (__bf16)b.z, (__bf16)b.w};
  *(uint4*)(dst + i) = *(uint4*)tmp;
}

// f32 [rows][cols] -> bf16 transposed [cols][rows]
__global__ void k_transpose_bf16(const float* __restrict__ src, __bf16* __restrict__ dst,
                                 int rows, int cols) {
  size_t i = (size_t)blockIdx.x * blockDim.x + threadIdx.x;
  if (i >= (size_t)rows * cols) return;
  int r = (int)(i / cols), c = (int)(i % cols);
  dst[(size_t)c * rows + r] = (__bf16)src[i];
}

// Pack one CfC layer: [w1|w2|wa|wb] ([cat,h] each) -> bf16 transposed padded
// wt[col][catP], col in [0,4*HP); padded f32 biases biasp[4*HP].
__global__ void k_pack_layer(const float* __restrict__ w1, const float* __restrict__ w2,
                             const float* __restrict__ wa, const float* __restrict__ wb,
                             const float* __restrict__ b1, const float* __restrict__ b2,
                             const float* __restrict__ ba, const float* __restrict__ bb,
                             __bf16* __restrict__ wt, float* __restrict__ biasp,
                             int seg0_len, int seg0_pad, int h, int catP, int HP) {
  size_t idx = (size_t)blockIdx.x * blockDim.x + threadIdx.x;
  size_t n_el = (size_t)4 * HP * catP;
  if (idx < (size_t)4 * HP) {
    int blk = (int)(idx / HP), c = (int)(idx % HP);
    const float* Bv = (blk == 0) ? b1 : (blk == 1) ? b2 : (blk == 2) ? ba : bb;
    biasp[idx] = (c < h) ? Bv[c] : 0.f;
  }
  if (idx >= n_el) return;
  int col = (int)(idx / catP);
  int k = (int)(idx % catP);
  int blk = col / HP, c = col % HP;
  int r = -1;
  if (k < seg0_len) r = k;
  else if (k >= seg0_pad && k < seg0_pad + h) r = seg0_len + (k - seg0_pad);
  float v = 0.f;
  if (r >= 0 && c < h) {
    const float* W = (blk == 0) ? w1 : (blk == 1) ? w2 : (blk == 2) ? wa : wb;
    v = W[(size_t)r * h + c];
  }
  wt[idx] = (__bf16)v;
}

// ---------------------------------------------------------------------------
// proj = x_bf16 @ w_in + b_in -> bf16 [65536][512]
// WG tile 128x64, 8 waves (4Mx2N). A tiles double-buffered in LDS via
// async-to-LDS copies (s_wait_asynccnt pipelining); B streamed from L2.
// ---------------------------------------------------------------------------
__global__ __launch_bounds__(256) void k_proj(const __bf16* __restrict__ xb,
                                              const __bf16* __restrict__ wt,
                                              const float* __restrict__ b_in,
                                              __bf16* __restrict__ proj) {
  __shared__ __bf16 ldsA[2][128 * 32];
  const int tid = threadIdx.x, lane = tid & 31, wv = tid >> 5;
  const int wm = wv >> 1, wn = wv & 1;
  const size_t r0 = (size_t)blockIdx.x * 128;
  const int c0 = blockIdx.y * 64;
  const int cr = tid >> 1;            // row 0..127
  const int cc = (tid & 1) * 16;      // col 0 / 16

  v8f acc[2][2];
#pragma unroll
  for (int i = 0; i < 2; ++i)
#pragma unroll
    for (int j = 0; j < 2; ++j) acc[i][j] = zero8();

  // prologue: stage k-tile 0
  copy16_g2l(xb + (r0 + cr) * IN_DIM + 0 + cc, &ldsA[0][cr * 32 + cc]);

  for (int kk = 0; kk < IN_DIM / 32; ++kk) {
    if (kk + 1 < IN_DIM / 32) {
      copy16_g2l(xb + (r0 + cr) * IN_DIM + (kk + 1) * 32 + cc,
                 &ldsA[(kk + 1) & 1][cr * 32 + cc]);
      wait_async2();   // tile kk landed (in-order completion), kk+1 in flight
    } else {
      wait_async0();
    }
    __syncthreads();
    const __bf16* at = ldsA[kk & 1];
    const int k0 = kk * 32;
#pragma unroll
    for (int i = 0; i < 2; ++i) {
      v16bf a = load_afrag(at + (wm * 32 + i * 16) * 32, 32, 0);
#pragma unroll
      for (int j = 0; j < 2; ++j) {
        const __bf16* colp = wt + (size_t)(c0 + wn * 32 + j * 16 + (lane & 15)) * IN_DIM;
        v16bf b = load_bfrag(colp, k0);
        acc[i][j] = wmma_bf16(a, b, acc[i][j]);
      }
    }
    __syncthreads();
  }
  const int mB = (lane >> 4) * 8;
#pragma unroll
  for (int i = 0; i < 2; ++i)
#pragma unroll
    for (int j = 0; j < 2; ++j) {
      const int n = c0 + wn * 32 + j * 16 + (lane & 15);
      const float bv = b_in[n];
#pragma unroll
      for (int v = 0; v < 8; ++v)
        proj[(r0 + wm * 32 + i * 16 + mB + v) * H_DIM + n] = (__bf16)(acc[i][j][v] + bv);
    }
}

// ---------------------------------------------------------------------------
// Persistent CfC scan: 4 WGs x 16 batch rows, 512 threads (16 waves).
// xt tiles are TDM-prefetched one timestep ahead (double-buffered staging);
// state kept in LDS; packed weights streamed from L2.
// ---------------------------------------------------------------------------
template <int CATP, int HP>
__device__ __forceinline__ void cfc_compute(const __bf16* xx, const __bf16* wt,
                                            int ntile, v8f acc[4]) {
  const int lane = threadIdx.x & 31;
  const __bf16* colp[4];
#pragma unroll
  for (int blk = 0; blk < 4; ++blk)
    colp[blk] = wt + (size_t)(blk * HP + ntile * 16 + (lane & 15)) * CATP;
#pragma unroll
  for (int blk = 0; blk < 4; ++blk) acc[blk] = zero8();
  for (int k0 = 0; k0 < CATP; k0 += 32) {
    v16bf a = load_afrag(xx, CATP, k0);
#pragma unroll
    for (int blk = 0; blk < 4; ++blk) {
      v16bf b = load_bfrag(colp[blk], k0);
      acc[blk] = wmma_bf16(a, b, acc[blk]);
    }
  }
}

__global__ __launch_bounds__(512) void k_scan(
    const __bf16* __restrict__ proj, const __bf16* __restrict__ w0t,
    const __bf16* __restrict__ w1t, const __bf16* __restrict__ w2t,
    const float* __restrict__ bias0, const float* __restrict__ bias1,
    const float* __restrict__ bias2, __bf16* __restrict__ cfc) {
  __shared__ __bf16 xx0[16 * CATP0];
  __shared__ __bf16 xx1[16 * CATP1];
  __shared__ __bf16 xx2[16 * CATP2];
  const int tid = threadIdx.x, lane = tid & 31, wv = tid >> 5;
  const int m0 = blockIdx.x * 16;

  for (int i = tid; i < 16 * CATP0; i += 512) xx0[i] = (__bf16)0.f;
  for (int i = tid; i < 16 * CATP1; i += 512) xx1[i] = (__bf16)0.f;
  for (int i = tid; i < 16 * CATP2; i += 512) xx2[i] = (__bf16)0.f;

#if HAVE_TDM
  __shared__ __bf16 xts[2][16 * H_DIM];   // TDM staging, double buffered
  if (wv == 0)
    tdm_load_tile(proj + (size_t)m0 * T_STEPS * H_DIM, lds_addr_of(&xts[0][0]),
                  H_DIM, 16, (unsigned long long)T_STEPS * H_DIM);
#endif
  if (__builtin_amdgcn_cluster_id_x() != 0) __builtin_amdgcn_s_cluster_barrier();
  __syncthreads();

  v8f acc[4];
  const int mB = (lane >> 4) * 8;

  for (int t = 0; t < T_STEPS; ++t) {
#if HAVE_TDM
    if (wv == 0) {
      if (t + 1 < T_STEPS) {
        tdm_load_tile(proj + ((size_t)m0 * T_STEPS + (t + 1)) * H_DIM,
                      lds_addr_of(&xts[(t + 1) & 1][0]), H_DIM, 16,
                      (unsigned long long)T_STEPS * H_DIM);
        __builtin_amdgcn_s_wait_tensorcnt(1);   // tile t done, t+1 in flight
      } else {
        __builtin_amdgcn_s_wait_tensorcnt(0);
      }
    }
    __syncthreads();
    {  // staged xt -> xx0 (wave = row, lane = 32B)
      const int r = wv, c = lane * 16;
      const uint4* s = (const uint4*)&xts[t & 1][r * H_DIM + c];
      uint4* d = (uint4*)&xx0[r * CATP0 + c];
      d[0] = s[0];
      d[1] = s[1];
    }
#else
    {  // direct xt fill
      const int r = wv, c = lane * 16;
      const uint4* s = (const uint4*)(proj + ((size_t)(m0 + r) * T_STEPS + t) * H_DIM + c);
      uint4* d = (uint4*)&xx0[r * CATP0 + c];
      d[0] = s[0];
      d[1] = s[1];
      if (t + 1 < T_STEPS)
        __builtin_prefetch(proj + ((size_t)(m0 + r) * T_STEPS + (t + 1)) * H_DIM + c, 0, 1);
    }
#endif
    __syncthreads();

    // ---- Layer 0: xx0[16x672] -> h0[16x160], 10 n-tiles ----
    if (wv < 10) cfc_compute<CATP0, HP0>(xx0, w0t, wv, acc);
    __syncthreads();
    if (wv < 10) {
      const int n = wv * 16 + (lane & 15);
#pragma unroll
      for (int v = 0; v < 8; ++v) {
        const float p1 = acc[0][v] + bias0[n];
        const float p2 = acc[1][v] + bias0[HP0 + n];
        const float s = sigmoidf_(acc[2][v] + bias0[2 * HP0 + n] + acc[3][v] + bias0[3 * HP0 + n]);
        const float h = tanhf(p1) * (1.f - s) + s * tanhf(p2);
        const __bf16 hb = (__bf16)h;
        const int m = mB + v;
        xx0[m * CATP0 + 512 + n] = hb;  // recurrent state
        xx1[m * CATP1 + n] = hb;        // forward to layer 1
      }
    }
    __syncthreads();

    // ---- Layer 1: xx1[16x288] -> h1[16x112], 7 n-tiles ----
    if (wv < 7) cfc_compute<CATP1, HP1>(xx1, w1t, wv, acc);
    __syncthreads();
    if (wv < 7) {
      const int n = wv * 16 + (lane & 15);
#pragma unroll
      for (int v = 0; v < 8; ++v) {
        const float p1 = acc[0][v] + bias1[n];
        const float p2 = acc[1][v] + bias1[HP1 + n];
        const float s = sigmoidf_(acc[2][v] + bias1[2 * HP1 + n] + acc[3][v] + bias1[3 * HP1 + n]);
        const float h = tanhf(p1) * (1.f - s) + s * tanhf(p2);
        const __bf16 hb = (__bf16)h;
        const int m = mB + v;
        xx1[m * CATP1 + 160 + n] = hb;  // recurrent state
        xx2[m * CATP2 + n] = hb;        // forward to layer 2
      }
    }
    __syncthreads();

    // ---- Layer 2: xx2[16x384] -> h2[16x256], 16 n-tiles (all waves) ----
    cfc_compute<CATP2, HP2>(xx2, w2t, wv, acc);
    __syncthreads();
    {
      const int n = wv * 16 + (lane & 15);
#pragma unroll
      for (int v = 0; v < 8; ++v) {
        const float p1 = acc[0][v] + bias2[n];
        const float p2 = acc[1][v] + bias2[HP2 + n];
        const float s = sigmoidf_(acc[2][v] + bias2[2 * HP2 + n] + acc[3][v] + bias2[3 * HP2 + n]);
        const float h = tanhf(p1) * (1.f - s) + s * tanhf(p2);
        const __bf16 hb = (__bf16)h;
        const int m = mB + v;
        xx2[m * CATP2 + 112 + n] = hb;  // recurrent state
        cfc[((size_t)(m0 + m) * T_STEPS + t) * OUT_DIM + n] = hb;  // motor output
      }
    }
    __syncthreads();
  }
}

// ---------------------------------------------------------------------------
// Head: out = gelu(cfc @ hw1 + hb1) @ hw2 + hb2, fused per 64-row tile.
// 8 waves (4Mx2N); hid (64x512 bf16) kept in LDS between the two GEMMs.
// ---------------------------------------------------------------------------
__global__ __launch_bounds__(256) void k_head(const __bf16* __restrict__ cfc,
                                              const __bf16* __restrict__ hw1t,
                                              const float* __restrict__ hb1,
                                              const __bf16* __restrict__ hw2t,
                                              const float* __restrict__ hb2,
                                              float* __restrict__ out) {
  __shared__ __bf16 hid[64 * H_DIM];
  const int tid = threadIdx.x, lane = tid & 31, wv = tid >> 5;
  const int wm = wv & 3, wn = wv >> 2;
  const size_t r0 = (size_t)blockIdx.x * 64;
  const int mB = (lane >> 4) * 8;

  // GEMM1 + exact-erf GELU: [64,256] @ [256,512]
  const __bf16* abase = cfc + (r0 + wm * 16) * OUT_DIM;
  for (int nt = wn * 16; nt < wn * 16 + 16; ++nt) {
    v8f acc = zero8();
    const __bf16* colp = hw1t + (size_t)(nt * 16 + (lane & 15)) * OUT_DIM;
    for (int k0 = 0; k0 < OUT_DIM; k0 += 32) {
      v16bf a = load_afrag(abase, OUT_DIM, k0);
      v16bf b = load_bfrag(colp, k0);
      acc = wmma_bf16(a, b, acc);
    }
    const int n = nt * 16 + (lane & 15);
    const float bv = hb1[n];
#pragma unroll
    for (int v = 0; v < 8; ++v) {
      const float xv = acc[v] + bv;
      const float g = 0.5f * xv * (1.f + erff(xv * 0.70710678118f));
      hid[(wm * 16 + mB + v) * H_DIM + n] = (__bf16)g;
    }
  }
  __syncthreads();

  // GEMM2: [64,512] @ [512,256] -> out (f32)
  for (int nt = wn * 8; nt < wn * 8 + 8; ++nt) {
    v8f acc = zero8();
    const __bf16* colp = hw2t + (size_t)(nt * 16 + (lane & 15)) * H_DIM;
    for (int k0 = 0; k0 < H_DIM; k0 += 32) {
      v16bf a = load_afrag(hid + (wm * 16) * H_DIM, H_DIM, k0);
      v16bf b = load_bfrag(colp, k0);
      acc = wmma_bf16(a, b, acc);
    }
    const int n = nt * 16 + (lane & 15);
    const float bv = hb2[n];
#pragma unroll
    for (int v = 0; v < 8; ++v)
      out[(r0 + wm * 16 + mB + v) * OUT_DIM + n] = acc[v] + bv;
  }
}

// ---------------------------------------------------------------------------
extern "C" void kernel_launch(void* const* d_in, const int* in_sizes, int n_in,
                              void* d_out, int out_size, void* d_ws, size_t ws_size,
                              hipStream_t stream) {
  const float* x    = (const float*)d_in[0];
  const float* w_in = (const float*)d_in[1];
  const float* b_in = (const float*)d_in[2];
  const float* L[3][8];
  for (int i = 0; i < 3; ++i)
    for (int j = 0; j < 8; ++j) L[i][j] = (const float*)d_in[3 + i * 8 + j];
  const float* hw1 = (const float*)d_in[27];
  const float* hb1 = (const float*)d_in[28];
  const float* hw2 = (const float*)d_in[29];
  const float* hb2 = (const float*)d_in[30];

  char* ws = (char*)d_ws;
  size_t off = 0;
  auto alloc = [&](size_t bytes) -> void* {
    void* p = ws + off;
    off += (bytes + 255) & ~(size_t)255;
    return p;
  };
  __bf16* x_bf16 = (__bf16*)alloc((size_t)M_ROWS * IN_DIM * 2);
  __bf16* w_in_t = (__bf16*)alloc((size_t)H_DIM * IN_DIM * 2);
  __bf16* proj   = (__bf16*)alloc((size_t)M_ROWS * H_DIM * 2);
  __bf16* w0t    = (__bf16*)alloc((size_t)4 * HP0 * CATP0 * 2);
  __bf16* w1t    = (__bf16*)alloc((size_t)4 * HP1 * CATP1 * 2);
  __bf16* w2t    = (__bf16*)alloc((size_t)4 * HP2 * CATP2 * 2);
  float* bias0   = (float*)alloc((size_t)4 * HP0 * 4);
  float* bias1   = (float*)alloc((size_t)4 * HP1 * 4);
  float* bias2   = (float*)alloc((size_t)4 * HP2 * 4);
  __bf16* cfc    = (__bf16*)alloc((size_t)M_ROWS * OUT_DIM * 2);
  __bf16* hw1t   = (__bf16*)alloc((size_t)H_DIM * OUT_DIM * 2);
  __bf16* hw2t   = (__bf16*)alloc((size_t)OUT_DIM * H_DIM * 2);

  // Weight / activation preparation
  k_cvt_bf16<<<(M_ROWS * (size_t)IN_DIM) / 8 / 256, 256, 0, stream>>>(
      x, x_bf16, (size_t)M_ROWS * IN_DIM);
  k_transpose_bf16<<<2048, 256, 0, stream>>>(w_in, w_in_t, IN_DIM, H_DIM);
  k_transpose_bf16<<<512, 256, 0, stream>>>(hw1, hw1t, OUT_DIM, H_DIM);
  k_transpose_bf16<<<512, 256, 0, stream>>>(hw2, hw2t, H_DIM, OUT_DIM);
  k_pack_layer<<<(4 * HP0 * CATP0 + 255) / 256, 256, 0, stream>>>(
      L[0][0], L[0][1], L[0][2], L[0][3], L[0][4], L[0][5], L[0][6], L[0][7],
      w0t, bias0, 512, 512, 154, CATP0, HP0);
  k_pack_layer<<<(4 * HP1 * CATP1 + 255) / 256, 256, 0, stream>>>(
      L[1][0], L[1][1], L[1][2], L[1][3], L[1][4], L[1][5], L[1][6], L[1][7],
      w1t, bias1, 154, 160, 102, CATP1, HP1);
  k_pack_layer<<<(4 * HP2 * CATP2 + 255) / 256, 256, 0, stream>>>(
      L[2][0], L[2][1], L[2][2], L[2][3], L[2][4], L[2][5], L[2][6], L[2][7],
      w2t, bias2, 102, 112, 256, CATP2, HP2);

  // Pipeline
  k_proj<<<dim3(M_ROWS / 128, H_DIM / 64), 256, 0, stream>>>(x_bf16, w_in_t, b_in, proj);
  k_scan<<<B_DIM / 16, 512, 0, stream>>>(proj, w0t, w1t, w2t, bias0, bias1, bias2, cfc);
  k_head<<<M_ROWS / 64, 256, 0, stream>>>(cfc, hw1t, hb1, hw2t, hb2, (float*)d_out);
}
